// HierarchicalAttention_3444563772076
// MI455X (gfx1250) — matrix operand order repaired and checked by reading, not compiled, not run on hardware
//
#include <hip/hip_runtime.h>
#include <hip/hip_bf16.h>
#include <cstdint>

// ---------------------------------------------------------------------------
// HierarchicalAttention on gfx1250 (MI455X).
//   scores = scale * q . (k + 0.2*bar_k + 0.1*beat_k)  -> one effective K
//   Phase 0: fp32 -> f16 conversion of x and the 6 weight matrices (once).
//   Phase 1: WMMA f16 GEMMs (q, k_eff via 3 accumulating launches, v^T).
//   Phase 2: causal flash attention, async-double-buffered K/V tiles.
//   Phase 3: output WMMA GEMM -> fp32 d_out.
// LDS tile fills use global_load_async_to_lds_b128 (ASYNCcnt) two batches
// deep; fragment feeds are 16B-contiguous so they lower to ds_load_b128.
// Gather indices / row base pointers are hoisted out of the K pipeline.
// ---------------------------------------------------------------------------

#define BB 4
#define SS 1024
#define DD 1024
#define HH 16
#define HD 64

typedef __attribute__((ext_vector_type(16))) _Float16 v16h;
typedef __attribute__((ext_vector_type(8)))  float    v8f;

union AFrag { v16h h; unsigned u[8]; };
union CFrag { v8f  v; float    f[8]; };

__device__ __forceinline__ unsigned pack2(float a, float b) {
    union { unsigned u; _Float16 h[2]; } t;
    t.h[0] = (_Float16)a; t.h[1] = (_Float16)b;
    return t.u;
}

// A-fragment (16x32, MxK) K-pair base for vgpr slot v (ISA 16-bit A table).
__device__ __forceinline__ int a_kbase(int v, int half) {
    return ((v < 4) ? 0 : 16) + half * 8 + (v & 3) * 2;
}
// B-fragment (32x16, KxN) K-pair base (same table order as A).
__device__ __forceinline__ int b_kbase(int v, int half) {
    return ((v < 4) ? 0 : 16) + half * 8 + (v & 3) * 2;
}

// ---- CDNA5 async global->LDS copy (ASYNCcnt), inline asm: portable across
// ---- ROCm 7.2 and amdgpu-toolchain (bypasses clang builtin arity skew).
__device__ __forceinline__ void async_b128(const void* g, unsigned lds_byte) {
    asm volatile("global_load_async_to_lds_b128 %0, %1, off"
                 :: "v"(lds_byte), "v"((unsigned long long)(uintptr_t)g)
                 : "memory");
}
template <int N>
__device__ __forceinline__ void wait_async() {
    asm volatile("s_wait_asynccnt %0" :: "i"(N) : "memory");
}
__device__ __forceinline__ unsigned lds_off(const void* p) {
    return (unsigned)(uintptr_t)p;  // low 32 bits of generic addr == LDS offset
}

// ---------------------------------------------------------------------------
// fp32 -> f16 bulk conversion (vectorized x4)
// ---------------------------------------------------------------------------
__global__ __launch_bounds__(256)
void cvt_f16(const float* __restrict__ s, _Float16* __restrict__ d, int n4) {
    int i = blockIdx.x * 256 + threadIdx.x;
    if (i < n4) {
        float4 f = ((const float4*)s)[i];
        uint2 o;
        o.x = pack2(f.x, f.y);
        o.y = pack2(f.z, f.w);
        ((uint2*)d)[i] = o;
    }
}

// ---------------------------------------------------------------------------
// proj_gemm: C = scaleMul * (A @ W^T + bias), A,W f16, K-step 64,
// 128x128 macro tile, 8 waves x (32x64), async double-buffered LDS tiles.
//   gather: optional int64 row gather along S (bar/beat keys).
//   mode 0: OutF[m*D+n] (f32, final projection)
//   mode 1: OutH [B,H,S,hd] f16 (q / k_eff; accumulate supported)
//   mode 2: OutH [B,H,hd,S] f16 (v transposed for attention B-frags)
// ---------------------------------------------------------------------------
#define GLD 72           // padded LDS row stride (halves): 144B, 16B-aligned
#define GTILE (128 * GLD)

__global__ __launch_bounds__(256)
void proj_gemm(const _Float16* __restrict__ A, const long long* __restrict__ gather,
               const _Float16* __restrict__ W, const float* __restrict__ bias,
               float* __restrict__ OutF, _Float16* __restrict__ OutH,
               float scaleMul, int accumulate, int mode)
{
    __shared__ __align__(16) _Float16 sAB[2 * 2 * GTILE];  // [buf][A|B]

    const int tid  = threadIdx.x;
    const int lane = tid & 31;
    const int w    = tid >> 5;
    const int half = lane >> 4;
    const int l15  = lane & 15;
    const int wrow = w >> 1;   // 0..3 -> 32-row band
    const int wcol = w & 1;    // 0..1 -> 64-col band

    const int m0 = blockIdx.y * 128;
    const int n0 = blockIdx.x * 128;
    const unsigned sbase = lds_off(sAB);

    CFrag acc[2][4];
#pragma unroll
    for (int mt = 0; mt < 2; ++mt)
#pragma unroll
        for (int nt = 0; nt < 4; ++nt)
#pragma unroll
            for (int r = 0; r < 8; ++r) acc[mt][nt].f[r] = 0.0f;

    // ---- hoist per-thread source row bases + LDS targets (K-invariant) ----
    const _Float16* srcA[4];
    const _Float16* srcB[4];
    unsigned dstOff[4];
#pragma unroll
    for (int i = 0; i < 4; ++i) {
        int c   = tid + i * 256;        // 0..1023 chunks of 16B
        int row = c >> 3;               // 0..127
        int seg = (c & 7) * 8;          // halves within 64-wide K slab
        int m  = m0 + row;
        int ss = m & (SS - 1);
        if (gather) ss = (int)gather[ss];
        srcA[i] = A + ((size_t)((m >> 10) * SS + ss) * DD + seg);
        srcB[i] = W + ((size_t)(n0 + row) * DD + seg);
        dstOff[i] = (unsigned)(row * GLD + seg) * 2;
    }

    // issue one 128x64 A tile + 128x64 B tile: 8 async b128 per thread
    auto issue = [&](int buf, int k0) {
        unsigned ldsA = sbase + (unsigned)(buf * 2 * GTILE) * 2;
        unsigned ldsB = ldsA + (unsigned)GTILE * 2;
#pragma unroll
        for (int i = 0; i < 4; ++i) {
            async_b128(srcA[i] + k0, ldsA + dstOff[i]);
            async_b128(srcB[i] + k0, ldsB + dstOff[i]);
        }
    };

    issue(0, 0);
    issue(1, 64);

    for (int kk = 0; kk < DD / 64; ++kk) {
        if (kk < DD / 64 - 1) wait_async<8>(); else wait_async<0>();
        __syncthreads();

        const _Float16* sA = sAB + (kk & 1) * 2 * GTILE;
        const _Float16* sB = sA + GTILE;
#pragma unroll
        for (int kh = 0; kh < 2; ++kh) {
            AFrag a[2], b[4];
#pragma unroll
            for (int mt = 0; mt < 2; ++mt) {
                int m = wrow * 32 + mt * 16 + l15;
#pragma unroll
                for (int v = 0; v < 8; ++v)
                    a[mt].u[v] = *(const unsigned*)&sA[m * GLD + kh * 32 + a_kbase(v, half)];
            }
#pragma unroll
            for (int nt = 0; nt < 4; ++nt) {
                int n = wcol * 64 + nt * 16 + l15;
#pragma unroll
                for (int v = 0; v < 8; ++v)
                    b[nt].u[v] = *(const unsigned*)&sB[n * GLD + kh * 32 + b_kbase(v, half)];
            }
#pragma unroll
            for (int mt = 0; mt < 2; ++mt)
#pragma unroll
                for (int nt = 0; nt < 4; ++nt)
                    acc[mt][nt].v = __builtin_amdgcn_wmma_f32_16x16x32_f16(
                        false, a[mt].h, false, b[nt].h, (short)0, acc[mt][nt].v,
                        false, false);
        }
        __syncthreads();
        int kn = kk + 2;
        if (kn < DD / 64) issue(kn & 1, kn * 64);
    }

    // ---- epilogue ----
#pragma unroll
    for (int mt = 0; mt < 2; ++mt) {
#pragma unroll
        for (int nt = 0; nt < 4; ++nt) {
            int n = n0 + wcol * 64 + nt * 16 + l15;
            float bv = bias[n];
#pragma unroll
            for (int r = 0; r < 8; ++r) {
                int m = m0 + wrow * 32 + mt * 16 + r + half * 8;
                float val = (acc[mt][nt].f[r] + bv) * scaleMul;
                int bb = m >> 10, ss = m & (SS - 1);
                int hh = n >> 6,  dd = n & 63;
                if (mode == 0) {
                    OutF[(size_t)m * DD + n] = val;
                } else if (mode == 1) {
                    size_t idx = (((size_t)bb * HH + hh) * SS + ss) * HD + dd;
                    OutH[idx] = accumulate ? (_Float16)((float)OutH[idx] + val)
                                           : (_Float16)val;
                } else {  // mode 2: [B,H,hd,S]
                    size_t idx = (((size_t)bb * HH + hh) * HD + dd) * SS + ss;
                    OutH[idx] = (_Float16)val;
                }
            }
        }
    }
}

// ---------------------------------------------------------------------------
// flash_attn: causal softmax(scale * Q Keff^T) V
//   Q,Keff f16 [B,H,S,64]; V f16 transposed [B,H,64,S]; O f16 [B,S,D] merged.
// Grid (S/128, B*H). 8 waves x 16-row Q strip; key blocks of 32, async
// double-buffered K/V^T tiles.
// ---------------------------------------------------------------------------
#define LDK 72   // K tile row stride (halves): 144B
#define LDV 40   // V^T tile row stride (halves): 80B
#define LDP 34
#define KVT (32 * LDK + 64 * LDV)   // halves per buffer

__global__ __launch_bounds__(256)
void flash_attn(const _Float16* __restrict__ Q, const _Float16* __restrict__ K,
                const _Float16* __restrict__ Vt, _Float16* __restrict__ O)
{
    __shared__ __align__(16) _Float16 sKV[2 * KVT];
    __shared__ __align__(16) _Float16 sP [8 * 16 * LDP];

    const int tid  = threadIdx.x;
    const int lane = tid & 31;
    const int w    = tid >> 5;
    const int half = lane >> 4;
    const int l15  = lane & 15;

    const int bh = blockIdx.y;
    const int q0 = blockIdx.x * 128;
    const size_t base  = (size_t)bh * SS * HD;   // Q/K rows
    const float scale = 0.125f;                  // 1/sqrt(64)
    const unsigned skv = lds_off(sKV);

    // ---- Q fragments: 16 rows x 64 d -> two 16x32 A-frags, direct f16 ----
    AFrag aQ[2];
    {
        const _Float16* qp = Q + base + (size_t)(q0 + w * 16 + l15) * HD;
#pragma unroll
        for (int dc = 0; dc < 2; ++dc)
#pragma unroll
            for (int v = 0; v < 8; ++v)
                aQ[dc].u[v] = *(const unsigned*)(qp + dc * 32 + a_kbase(v, half));
    }

    float mrow[8], lrow[8];
    CFrag o[4];
#pragma unroll
    for (int r = 0; r < 8; ++r) { mrow[r] = -1e30f; lrow[r] = 0.0f; }
#pragma unroll
    for (int dc = 0; dc < 4; ++dc)
#pragma unroll
        for (int r = 0; r < 8; ++r) o[dc].f[r] = 0.0f;

    // ---- hoist per-thread K / V^T source bases + LDS targets (j-invariant) --
    const int kkey = tid >> 3, kseg = (tid & 7) * 8;   // K tile: 256 chunks
    const int vd   = tid >> 2, vseg = (tid & 3) * 8;   // V^T tile: 256 chunks
    const _Float16* srcK = K + base + (size_t)kkey * HD + kseg;
    const _Float16* srcV = Vt + (size_t)bh * HD * SS + (size_t)vd * SS + vseg;
    const unsigned dstK = (unsigned)(kkey * LDK + kseg) * 2;
    const unsigned dstV = (unsigned)(32 * LDK + vd * LDV + vseg) * 2;

    // issue K (32x64) + V^T (64x32) tile: 2 async b128 per thread
    auto issue_kv = [&](int buf, int j0) {
        unsigned dst = skv + (unsigned)(buf * KVT) * 2;
        async_b128(srcK + (size_t)j0 * HD, dst + dstK);
        async_b128(srcV + j0,              dst + dstV);
    };

    _Float16* sPw = &sP[w * 16 * LDP];
    const int nIter = (q0 + 128) / 32;

    issue_kv(0, 0);
    issue_kv(1, 32);

    for (int jj = 0; jj < nIter; ++jj) {
        if (jj < nIter - 1) wait_async<2>(); else wait_async<0>();
        __syncthreads();

        const _Float16* sK  = sKV + (jj & 1) * KVT;
        const _Float16* sVt = sK + 32 * LDK;
        const int j0 = jj * 32;

        // ---- S[16x32] = Q(16x64) . Keff(32x64)^T, 4 WMMAs ----
        CFrag s[2];
#pragma unroll
        for (int n = 0; n < 2; ++n) {
#pragma unroll
            for (int r = 0; r < 8; ++r) s[n].f[r] = 0.0f;
            int key = 16 * n + l15;
#pragma unroll
            for (int dc = 0; dc < 2; ++dc) {
                AFrag bK;
#pragma unroll
                for (int v = 0; v < 8; ++v)
                    bK.u[v] = *(const unsigned*)&sK[key * LDK + dc * 32 + b_kbase(v, half)];
                s[n].v = __builtin_amdgcn_wmma_f32_16x16x32_f16(
                    false, aQ[dc].h, false, bK.h, (short)0, s[n].v, false, false);
            }
            int jg = j0 + 16 * n + l15;
#pragma unroll
            for (int r = 0; r < 8; ++r) {
                int ig = q0 + w * 16 + r + half * 8;
                float x = s[n].f[r] * scale;
                s[n].f[r] = (jg > ig) ? -1e30f : x;  // causal mask
            }
        }

        // ---- online softmax: row reductions across each 16-lane half ----
        float alpha[8];
#pragma unroll
        for (int r = 0; r < 8; ++r) {
            float tm = fmaxf(s[0].f[r], s[1].f[r]);
#pragma unroll
            for (int off = 1; off < 16; off <<= 1)
                tm = fmaxf(tm, __shfl_xor(tm, off, 32));
            float mn = fmaxf(mrow[r], tm);
            alpha[r] = __expf(mrow[r] - mn);
            mrow[r] = mn;
            float p0 = __expf(s[0].f[r] - mn);
            float p1 = __expf(s[1].f[r] - mn);
            s[0].f[r] = p0; s[1].f[r] = p1;
            float rs = p0 + p1;
#pragma unroll
            for (int off = 1; off < 16; off <<= 1)
                rs += __shfl_xor(rs, off, 32);
            lrow[r] = lrow[r] * alpha[r] + rs;
        }
#pragma unroll
        for (int dc = 0; dc < 4; ++dc)
#pragma unroll
            for (int r = 0; r < 8; ++r) o[dc].f[r] *= alpha[r];

        // ---- re-fragment P: C-layout -> A-layout through LDS ----
#pragma unroll
        for (int n = 0; n < 2; ++n)
#pragma unroll
            for (int r = 0; r < 8; ++r)
                sPw[(r + half * 8) * LDP + 16 * n + l15] = (_Float16)s[n].f[r];
        __syncthreads();

        // ---- O += P(16x32) . V(32x64), 4 WMMAs ----
        AFrag aP;
#pragma unroll
        for (int v = 0; v < 8; ++v)
            aP.u[v] = *(const unsigned*)&sPw[l15 * LDP + a_kbase(v, half)];
#pragma unroll
        for (int dc = 0; dc < 4; ++dc) {
            AFrag bV;
            int d = dc * 16 + l15;
#pragma unroll
            for (int v = 0; v < 8; ++v)
                bV.u[v] = *(const unsigned*)&sVt[d * LDV + b_kbase(v, half)];
            o[dc].v = __builtin_amdgcn_wmma_f32_16x16x32_f16(
                false, aP.h, false, bV.h, (short)0, o[dc].v, false, false);
        }
        __syncthreads();  // buffer consumers done before refill

        int jn = jj + 2;
        if (jn < nIter) issue_kv(jn & 1, jn * 32);
    }

    // ---- normalize, write head-merged [B,S,D] f16 ----
    const int b = bh >> 4, h = bh & 15;
#pragma unroll
    for (int dc = 0; dc < 4; ++dc)
#pragma unroll
        for (int r = 0; r < 8; ++r) {
            int i = q0 + w * 16 + r + half * 8;
            int col = h * HD + dc * 16 + l15;
            O[((size_t)b * SS + i) * DD + col] = (_Float16)(o[dc].f[r] / lrow[r]);
        }
}

// ---------------------------------------------------------------------------
extern "C" void kernel_launch(void* const* d_in, const int* in_sizes, int n_in,
                              void* d_out, int out_size, void* d_ws, size_t ws_size,
                              hipStream_t stream) {
    (void)in_sizes; (void)n_in; (void)out_size; (void)ws_size;
    const float*     x    = (const float*)d_in[0];
    // d_in[1] = causal tril mask: implemented directly, not read
    const long long* bar  = (const long long*)d_in[2];
    const long long* beat = (const long long*)d_in[3];
    const float *Wq = (const float*)d_in[4],  *bq = (const float*)d_in[5];
    const float *Wk = (const float*)d_in[6],  *bk = (const float*)d_in[7];
    const float *Wv = (const float*)d_in[8],  *bv = (const float*)d_in[9];
    const float *Wbar = (const float*)d_in[10], *bbar = (const float*)d_in[11];
    const float *Wbt  = (const float*)d_in[12], *bbt  = (const float*)d_in[13];
    const float *Wo = (const float*)d_in[14], *bo = (const float*)d_in[15];

    const size_t XN = (size_t)BB * SS * DD;   // 4 Mi
    const size_t WN = (size_t)DD * DD;        // 1 Mi

    _Float16* p = (_Float16*)d_ws;
    _Float16* xh   = p; p += XN;
    _Float16* Wqh  = p; p += WN;
    _Float16* Wkh  = p; p += WN;
    _Float16* Wvh  = p; p += WN;
    _Float16* Wbrh = p; p += WN;
    _Float16* Wbth = p; p += WN;
    _Float16* Woh  = p; p += WN;
    _Float16* qh   = p; p += XN;
    _Float16* keff = p; p += XN;
    _Float16* vt   = p; p += XN;
    _Float16* aout = p; p += XN;

    // ---- phase 0: one-time fp32 -> f16 conversions ----
    cvt_f16<<<(int)(XN / 4 / 256), 256, 0, stream>>>(x, xh, (int)(XN / 4));
    cvt_f16<<<(int)(WN / 4 / 256), 256, 0, stream>>>(Wq,   Wqh,  (int)(WN / 4));
    cvt_f16<<<(int)(WN / 4 / 256), 256, 0, stream>>>(Wk,   Wkh,  (int)(WN / 4));
    cvt_f16<<<(int)(WN / 4 / 256), 256, 0, stream>>>(Wv,   Wvh,  (int)(WN / 4));
    cvt_f16<<<(int)(WN / 4 / 256), 256, 0, stream>>>(Wbar, Wbrh, (int)(WN / 4));
    cvt_f16<<<(int)(WN / 4 / 256), 256, 0, stream>>>(Wbt,  Wbth, (int)(WN / 4));
    cvt_f16<<<(int)(WN / 4 / 256), 256, 0, stream>>>(Wo,   Woh,  (int)(WN / 4));

    dim3 gG(DD / 128, (BB * SS) / 128);  // (8, 32)

    // ---- phase 1: projections ----
    proj_gemm<<<gG, 256, 0, stream>>>(xh, nullptr, Wqh, bq, nullptr, qh,   1.0f, 0, 1);
    proj_gemm<<<gG, 256, 0, stream>>>(xh, nullptr, Wvh, bv, nullptr, vt,   1.0f, 0, 2);
    proj_gemm<<<gG, 256, 0, stream>>>(xh, nullptr, Wkh,  bk,   nullptr, keff, 1.0f, 0, 1);
    proj_gemm<<<gG, 256, 0, stream>>>(xh, bar,     Wbrh, bbar, nullptr, keff, 0.2f, 1, 1);
    proj_gemm<<<gG, 256, 0, stream>>>(xh, beat,    Wbth, bbt,  nullptr, keff, 0.1f, 1, 1);

    // ---- phase 2: causal flash attention ----
    flash_attn<<<dim3(SS / 128, BB * HH), 256, 0, stream>>>(qh, keff, vt, aout);

    // ---- phase 3: output projection -> fp32 d_out ----
    proj_gemm<<<gG, 256, 0, stream>>>(aout, nullptr, Woh, bo, (float*)d_out, nullptr, 1.0f, 0, 0);
}
